// EncoderLayer_43147241456128
// MI455X (gfx1250) — matrix operand order repaired and checked by reading, not compiled, not run on hardware
//
#include <hip/hip_runtime.h>
#include <stdint.h>

typedef __attribute__((ext_vector_type(16))) _Float16 v16h;
typedef __attribute__((ext_vector_type(8)))  float    v8f;
typedef __attribute__((ext_vector_type(2)))  _Float16 h2;

#define B_    4
#define T_    512
#define E_    512
#define S_    1024
#define HID   512
#define HD    32
#define NHEAD 16
#define POS   3
#define D_    96          // head feature dim = POS*HD
#define SMOOTH 20.0f
#define LNEPS  1e-3f
#define LOG2E  1.44269504088896340736f
#define QSCALE 0.10206207261596575f   // sqrt(HD/3)/HD

#ifndef __has_builtin
#define __has_builtin(x) 0
#endif

// ---- CDNA5 async global->LDS path (guarded; sync fallback keeps build safe)
#if __has_builtin(__builtin_amdgcn_global_load_async_to_lds_b32)
#define HAVE_ASYNC_LDS 1
typedef __attribute__((address_space(1))) int gint_t;   // global (clang: '__device__ int *')
typedef __attribute__((address_space(3))) int lint_t;   // LDS
#if __has_builtin(__builtin_amdgcn_s_wait_asynccnt)
#define WAIT_ASYNC(n) __builtin_amdgcn_s_wait_asynccnt(n)
#else
#define WAIT_ASYNC(n) asm volatile("s_wait_asynccnt %0" ::"i"(n) : "memory")
#endif
#else
#define HAVE_ASYNC_LDS 0
#define WAIT_ASYNC(n) do { } while (0)
#endif

static constexpr int KP = 98;   // K/V tile pitch (halves): 49 dwords -> conflict free
static constexpr int PP = 34;   // P tile pitch: 17 dwords -> conflict free
static constexpr int WP = 34;   // W tile pitch

__device__ __forceinline__ void cp_b32(const _Float16* g, _Float16* l) {
#if HAVE_ASYNC_LDS
    __builtin_amdgcn_global_load_async_to_lds_b32((gint_t*)g, (lint_t*)l, 0, 0);
#else
    *(uint32_t*)l = *(const uint32_t*)g;
#endif
}

// -------- pack K/V to f16 head-major layout with PBC gather resolved ------
// kh/vh: (b, h, s, 96) halves, s in [0,S)
__global__ __launch_bounds__(256) void pack_kv(
    const float* __restrict__ k, const float* __restrict__ v,
    const int* __restrict__ idx, _Float16* __restrict__ kh,
    _Float16* __restrict__ vh)
{
    const int s = blockIdx.x, b = blockIdx.y;
    const int src = (s < T_) ? s : idx[b * E_ + (s - T_)];
    const float* kb = k + ((size_t)(b * T_ + src)) * POS * HID;
    const float* vb = v + ((size_t)(b * T_ + src)) * POS * HID;
    for (int e = threadIdx.x; e < POS * HID; e += 256) {
        int p = e >> 9, rem = e & 511;
        int hh = rem >> 5, c = rem & 31;
        size_t o = ((size_t)(b * NHEAD + hh) * S_ + s) * D_ + p * HD + c;
        kh[o] = (_Float16)kb[e];
        vh[o] = (_Float16)vb[e];
    }
}

// -------- pack projection weight to f16 -----------------------------------
__global__ __launch_bounds__(256) void pack_w(
    const float* __restrict__ w, _Float16* __restrict__ wh)
{
    int i = (blockIdx.x * 256 + (int)threadIdx.x) * 4;
    float4 wv = *(const float4*)(w + i);
    wh[i]     = (_Float16)wv.x;
    wh[i + 1] = (_Float16)wv.y;
    wh[i + 2] = (_Float16)wv.z;
    wh[i + 3] = (_Float16)wv.w;
}

// -------------------- flash attention (f16 WMMA, f32 softmax) -------------
__global__ __launch_bounds__(256) void attn_kernel(
    const float* __restrict__ q, const _Float16* __restrict__ kh,
    const _Float16* __restrict__ vh,
    const float* __restrict__ bias, const float* __restrict__ law,
    const uint8_t* __restrict__ kpm, const uint8_t* __restrict__ em,
    float* __restrict__ attn_out)
{
    __shared__ _Float16 Kt2[2][32 * KP];
    __shared__ _Float16 Vt2[2][32 * KP];
    __shared__ _Float16 Pb[8 * 16 * PP];

    const int b  = blockIdx.z;
    const int h  = blockIdx.y;
    const int wv = threadIdx.x >> 5;
    const int lane = threadIdx.x & 31;
    const int n16 = lane & 15;
    const int hs  = lane >> 4;
    const int t0  = blockIdx.x * 128 + wv * 16;

    const _Float16* kbase = kh + ((size_t)(b * NHEAD + h)) * S_ * D_;
    const _Float16* vbase = vh + ((size_t)(b * NHEAD + h)) * S_ * D_;

    // ---- Q A-fragments (16x32 f16 per fragment, 3 fragments for d=96) ----
    v16h a[3];
    #pragma unroll
    for (int f = 0; f < 3; ++f) {
        const float* qp = q + (((size_t)(b * T_ + t0 + n16)) * POS + f) * HID + h * HD;
        #pragma unroll
        for (int j = 0; j < 8; ++j) {
            int kk = 2 * j + 16 * hs;
            float2 qv = *(const float2*)(qp + kk);
            a[f][2 * j]     = (_Float16)(qv.x * QSCALE);
            a[f][2 * j + 1] = (_Float16)(qv.y * QSCALE);
        }
    }

    v8f o[6];
    #pragma unroll
    for (int nt = 0; nt < 6; ++nt)
        #pragma unroll
        for (int r = 0; r < 8; ++r) o[nt][r] = 0.0f;

    float mrun[8], lrun[8];
    #pragma unroll
    for (int r = 0; r < 8; ++r) { mrun[r] = -1e30f; lrun[r] = 0.0f; }

    // ---- prologue: stage chunk 0 (async: 12 b32 instrs per thread) -------
    for (int i = threadIdx.x; i < 32 * 48; i += 256) {
        int row = i / 48, wo = i - row * 48;
        cp_b32(kbase + 2 * i, &Kt2[0][row * KP + 2 * wo]);
        cp_b32(vbase + 2 * i, &Vt2[0][row * KP + 2 * wo]);
    }

    int buf = 0;
    constexpr int NIT = S_ / 32;
    for (int it = 0; it < NIT; ++it) {
        const int s0 = it * 32;

        // ---- issue async stage of next chunk into the other buffer ------
        if (it + 1 < NIT) {
            const _Float16* kg = kbase + (size_t)(s0 + 32) * D_;
            const _Float16* vg = vbase + (size_t)(s0 + 32) * D_;
            _Float16* kl = Kt2[buf ^ 1];
            _Float16* vl = Vt2[buf ^ 1];
            for (int i = threadIdx.x; i < 32 * 48; i += 256) {
                int row = i / 48, wo = i - row * 48;
                cp_b32(kg + 2 * i, &kl[row * KP + 2 * wo]);
                cp_b32(vg + 2 * i, &vl[row * KP + 2 * wo]);
            }
            WAIT_ASYNC(12);          // retire current chunk, leave next in flight
        } else {
            WAIT_ASYNC(0);
        }
        __syncthreads();             // all waves' current-chunk writes visible

        // cache hint for next bias block (global_prefetch_b8)
        if (lane < 16 && s0 + 32 < S_) {
            __builtin_prefetch(
                bias + (((size_t)(b * NHEAD + h)) * T_ + (t0 + lane)) * S_ + s0 + 32, 0, 1);
        }

        const _Float16* Kt = Kt2[buf];
        const _Float16* Vt = Vt2[buf];

        // ---- scores: two 16x16 tiles (columns s0..+15, s0+16..+31) -------
        v8f aw[2];
        float lawv[2][8];
        #pragma unroll
        for (int nh = 0; nh < 2; ++nh) {
            v8f scacc;
            #pragma unroll
            for (int r = 0; r < 8; ++r) scacc[r] = 0.0f;
            #pragma unroll
            for (int f = 0; f < 3; ++f) {
                v16h bfrag;
                const _Float16* kp2 = &Kt[(nh * 16 + n16) * KP + f * 32 + 16 * hs];
                #pragma unroll
                for (int j = 0; j < 8; ++j) {
                    h2 kv = *(const h2*)(kp2 + 2 * j);
                    bfrag[2 * j] = kv.x; bfrag[2 * j + 1] = kv.y;
                }
                scacc = __builtin_amdgcn_wmma_f32_16x16x32_f16(
                    false, a[f], false, bfrag, (short)0, scacc, false, false);
            }
            const int scol = s0 + nh * 16 + n16;
            const bool masked = (scol < T_) ? (kpm[b * T_ + scol] != 0)
                                            : (em[b * E_ + (scol - T_)] != 0);
            const float* bp = bias + (((size_t)(b * NHEAD + h)) * T_ + (t0 + 8 * hs)) * S_ + scol;
            const float* lp = law  + ((size_t)b * T_ + (t0 + 8 * hs)) * S_ + scol;
            #pragma unroll
            for (int r = 0; r < 8; ++r) {
                float lw = lp[(size_t)r * S_];
                float av = (scacc[r] + bp[(size_t)r * S_] + SMOOTH) * lw - SMOOTH;
                if (masked) av = -1e30f;
                aw[nh][r] = av;
                lawv[nh][r] = lw;
            }
        }

        // ---- online softmax update (16-lane group reductions) ------------
        float scale[8];
        #pragma unroll
        for (int r = 0; r < 8; ++r) {
            float tm = fmaxf(aw[0][r], aw[1][r]);
            #pragma unroll
            for (int off = 8; off >= 1; off >>= 1)
                tm = fmaxf(tm, __shfl_xor(tm, off, 16));
            float nm = fmaxf(mrun[r], tm);
            float sc = exp2f((mrun[r] - nm) * LOG2E);
            float p0 = exp2f((aw[0][r] - nm) * LOG2E);
            float p1 = exp2f((aw[1][r] - nm) * LOG2E);
            float ps = p0 + p1;
            #pragma unroll
            for (int off = 8; off >= 1; off >>= 1)
                ps += __shfl_xor(ps, off, 16);
            lrun[r] = lrun[r] * sc + ps;
            mrun[r] = nm;
            scale[r] = sc;
            aw[0][r] = p0 * lawv[0][r];   // probs *= law (numerator only)
            aw[1][r] = p1 * lawv[1][r];
        }
        #pragma unroll
        for (int nt = 0; nt < 6; ++nt)
            #pragma unroll
            for (int r = 0; r < 8; ++r) o[nt][r] *= scale[r];

        // ---- re-layout P (C-layout f32 -> A-layout f16) via wave LDS -----
        _Float16* pb = &Pb[(wv * 16) * PP];
        #pragma unroll
        for (int r = 0; r < 8; ++r) {
            pb[(r + 8 * hs) * PP + n16]      = (_Float16)aw[0][r];
            pb[(r + 8 * hs) * PP + 16 + n16] = (_Float16)aw[1][r];
        }
        v16h pa;
        const _Float16* pr = &Pb[(wv * 16 + n16) * PP];
        #pragma unroll
        for (int j = 0; j < 8; ++j) {
            int kk = 2 * j + 16 * hs;
            h2 pv = *(const h2*)(pr + kk);
            pa[2 * j] = pv.x; pa[2 * j + 1] = pv.y;
        }

        // ---- PV: accumulate into 6 output n-tiles ------------------------
        #pragma unroll
        for (int nt = 0; nt < 6; ++nt) {
            v16h bv;
            const _Float16* vp = &Vt[(16 * hs) * KP + nt * 16 + n16];
            #pragma unroll
            for (int j = 0; j < 8; ++j) {
                bv[2 * j]     = vp[(size_t)(2 * j) * KP];
                bv[2 * j + 1] = vp[(size_t)(2 * j + 1) * KP];
            }
            o[nt] = __builtin_amdgcn_wmma_f32_16x16x32_f16(
                false, pa, false, bv, (short)0, o[nt], false, false);
        }

        __syncthreads();             // done reading buf before it is restaged
        buf ^= 1;
    }

    // ---- epilogue: normalize, scatter to (b,t,p,HID) ---------------------
    float rl[8];
    #pragma unroll
    for (int r = 0; r < 8; ++r) rl[r] = 1.0f / lrun[r];
    #pragma unroll
    for (int nt = 0; nt < 6; ++nt) {
        int d = nt * 16 + n16;
        #pragma unroll
        for (int r = 0; r < 8; ++r) {
            size_t oidx = (((size_t)(b * T_ + t0 + r + 8 * hs)) * POS + (d >> 5)) * HID
                          + h * HD + (d & 31);
            attn_out[oidx] = o[nt][r] * rl[r];
        }
    }
}

// -------------------- equivariant LN inverse norm -------------------------
__global__ __launch_bounds__(256) void ln_inv_kernel(
    const float* __restrict__ attn, float* __restrict__ inv)
{
    const int bt = blockIdx.x;
    const float* x = attn + (size_t)bt * POS * HID;
    float s = 0.0f;
    for (int e = threadIdx.x; e < POS * HID; e += 256) {
        float xv = x[e];
        s += xv * xv;
    }
    __shared__ float red[8];
    #pragma unroll
    for (int off = 16; off >= 1; off >>= 1) s += __shfl_xor(s, off, 32);
    if ((threadIdx.x & 31) == 0) red[threadIdx.x >> 5] = s;
    __syncthreads();
    if (threadIdx.x == 0) {
        float t = 0.0f;
        #pragma unroll
        for (int i = 0; i < 8; ++i) t += red[i];
        inv[bt] = rsqrtf(t / (float)HID + LNEPS);
    }
}

// -------------------- out projection GEMM (f16 WMMA) ----------------------
// out[row, o] = sum_c (attn[row,c] * lnw[c] * inv[row/3]) * w[o, c]
__global__ __launch_bounds__(256) void proj_kernel(
    const float* __restrict__ attn, const float* __restrict__ inv,
    const float* __restrict__ lnw, const _Float16* __restrict__ wh,
    float* __restrict__ out)
{
    __shared__ _Float16 Wt2[2][64 * WP];
    const int nb = blockIdx.x, mb = blockIdx.y;
    const int wvi = threadIdx.x >> 5;
    const int lane = threadIdx.x & 31;
    const int n16 = lane & 15;
    const int hs  = lane >> 4;
    const int row0 = mb * 128 + wvi * 16;
    const int n0 = nb * 64;

    v8f acc[4];
    #pragma unroll
    for (int nt = 0; nt < 4; ++nt)
        #pragma unroll
        for (int r = 0; r < 8; ++r) acc[nt][r] = 0.0f;

    const int mrow = row0 + n16;
    const float invr = inv[mrow / 3];

    // prologue: stage W tile for c0=0 (4 async b32 per thread)
    for (int i = threadIdx.x; i < 64 * 16; i += 256) {
        int nn = i >> 4, wo = i & 15;
        cp_b32(wh + (size_t)(n0 + nn) * HID + 2 * wo, &Wt2[0][nn * WP + 2 * wo]);
    }

    int buf = 0;
    constexpr int NKI = HID / 32;
    for (int it = 0; it < NKI; ++it) {
        const int c0 = it * 32;
        if (it + 1 < NKI) {
            _Float16* wl = Wt2[buf ^ 1];
            for (int i = threadIdx.x; i < 64 * 16; i += 256) {
                int nn = i >> 4, wo = i & 15;
                cp_b32(wh + (size_t)(n0 + nn) * HID + (c0 + 32) + 2 * wo,
                       &wl[nn * WP + 2 * wo]);
            }
            WAIT_ASYNC(4);
        } else {
            WAIT_ASYNC(0);
        }
        __syncthreads();

        v16h af;
        const float* ap = attn + (size_t)mrow * HID + c0;
        #pragma unroll
        for (int j = 0; j < 8; ++j) {
            int kk = 2 * j + 16 * hs;
            float2 av = *(const float2*)(ap + kk);
            float2 lw = *(const float2*)(lnw + c0 + kk);
            af[2 * j]     = (_Float16)(av.x * lw.x * invr);
            af[2 * j + 1] = (_Float16)(av.y * lw.y * invr);
        }
        const _Float16* Wt = Wt2[buf];
        #pragma unroll
        for (int nt = 0; nt < 4; ++nt) {
            v16h bf;
            const _Float16* wp = &Wt[(nt * 16 + n16) * WP + 16 * hs];
            #pragma unroll
            for (int j = 0; j < 8; ++j) {
                h2 wv2 = *(const h2*)(wp + 2 * j);
                bf[2 * j] = wv2.x; bf[2 * j + 1] = wv2.y;
            }
            acc[nt] = __builtin_amdgcn_wmma_f32_16x16x32_f16(
                false, af, false, bf, (short)0, acc[nt], false, false);
        }
        __syncthreads();
        buf ^= 1;
    }

    #pragma unroll
    for (int nt = 0; nt < 4; ++nt)
        #pragma unroll
        for (int r = 0; r < 8; ++r)
            out[(size_t)(row0 + r + 8 * hs) * HID + n0 + nt * 16 + n16] = acc[nt][r];
}

// --------------------------------------------------------------------------
extern "C" void kernel_launch(void* const* d_in, const int* in_sizes, int n_in,
                              void* d_out, int out_size, void* d_ws, size_t ws_size,
                              hipStream_t stream) {
    const float*   q    = (const float*)d_in[0];
    const float*   k    = (const float*)d_in[1];
    const float*   v    = (const float*)d_in[2];
    const float*   bias = (const float*)d_in[3];
    const float*   law  = (const float*)d_in[4];
    const float*   lnw  = (const float*)d_in[5];
    const float*   w    = (const float*)d_in[6];
    const int*     idx  = (const int*)d_in[7];
    const uint8_t* kpm  = (const uint8_t*)d_in[8];
    const uint8_t* em   = (const uint8_t*)d_in[9];

    const size_t NATTN = (size_t)B_ * T_ * POS * HID;     // 3.1M floats
    const size_t NKH   = (size_t)B_ * NHEAD * S_ * D_;    // 6.3M halves

    float*    attn_ws = (float*)d_ws;
    float*    inv_ws  = attn_ws + NATTN;
    _Float16* kh      = (_Float16*)(inv_ws + (size_t)B_ * T_);
    _Float16* vh      = kh + NKH;
    _Float16* wh      = vh + NKH;
    float*    out     = (float*)d_out;

    pack_kv<<<dim3(S_, B_), 256, 0, stream>>>(k, v, idx, kh, vh);
    pack_w<<<dim3((HID * HID) / 1024), 256, 0, stream>>>(w, wh);
    attn_kernel<<<dim3(T_ / 128, NHEAD, B_), 256, 0, stream>>>(
        q, kh, vh, bias, law, kpm, em, attn_ws);
    ln_inv_kernel<<<dim3(B_ * T_), 256, 0, stream>>>(attn_ws, inv_ws);
    proj_kernel<<<dim3(HID / 64, (B_ * T_ * POS) / 128), 256, 0, stream>>>(
        attn_ws, inv_ws, lnw, wh, out);
}
